// CellTokenLLM_72782515798417
// MI455X (gfx1250) — matrix operand order repaired and checked by reading, not compiled, not run on hardware
//
#include <hip/hip_runtime.h>
#include <hip/hip_bf16.h>
#include <math.h>

// ---------------- model constants ----------------
#define CD_V     32000
#define CD_D     512
#define CD_NB    8
#define CD_STEPS 5
#define CD_H     8
#define CD_DEPTH 4
#define CD_S     2048
#define CD_B     2
#define CD_DH    64
#define CD_FF    2048
#define CD_ROWS  (CD_B * CD_S)   // 4096
#define CD_INNER 96              // NB*4 + NB*NB

#define GAS __attribute__((address_space(1)))

typedef __attribute__((ext_vector_type(16))) __bf16       bf16x16;
typedef __attribute__((ext_vector_type(8)))  float        f32x8;
typedef __attribute__((ext_vector_type(4)))  unsigned int u32x4;

typedef GAS const unsigned short* gusp;     // global-addrspace u16 pointer

union FragU { bf16x16 v; u32x4 q[2]; unsigned short s[16]; };

__device__ __forceinline__ gusp to_global(const unsigned short* p) {
  return (gusp)(unsigned long long)p;
}

__device__ __forceinline__ void load_frag(FragU& U, gusp p) {
  U.q[0] = ((GAS const u32x4*)p)[0];        // bytes 0..15
  U.q[1] = ((GAS const u32x4*)(p + 16))[0]; // bytes 32..47
}

__device__ __forceinline__ unsigned short f2bf(float f) {
  union { float f; unsigned u; } v; v.f = f;
  unsigned r = v.u + 0x7FFFu + ((v.u >> 16) & 1u);   // round-to-nearest-even
  return (unsigned short)(r >> 16);
}

// ---------------- fp32 -> bf16 conversion ----------------
__global__ void f2bf_kernel(const float* __restrict__ in,
                            unsigned short* __restrict__ out, long n) {
  long i = (long)blockIdx.x * blockDim.x + threadIdx.x;
  long stride = (long)gridDim.x * blockDim.x;
  for (; i < n; i += stride) out[i] = f2bf(in[i]);
}

// ---------------- embedding gather + stim ----------------
__global__ void embed_kernel(const int* __restrict__ tok,
                             const float* __restrict__ tokEmb,
                             const float* __restrict__ posEmb,
                             float* __restrict__ e,
                             unsigned short* __restrict__ ebf,
                             float* __restrict__ stim) {
  int row = blockIdx.x;                 // b*S + s
  int s   = row & (CD_S - 1);
  int t   = tok[row];
  __shared__ float red[256];
  float local = 0.f;
  for (int d = threadIdx.x; d < CD_D; d += 256) {
    float v = tokEmb[(size_t)t * CD_D + d] + posEmb[(size_t)s * CD_D + d];
    e[(size_t)row * CD_D + d]   = v;
    ebf[(size_t)row * CD_D + d] = f2bf(v);
    local += v * v;
  }
  red[threadIdx.x] = local;
  __syncthreads();
  for (int off = 128; off > 0; off >>= 1) {
    if (threadIdx.x < off) red[threadIdx.x] += red[threadIdx.x + off];
    __syncthreads();
  }
  if (threadIdx.x == 0) {
    float n = sqrtf(red[0]) * 0.0441941738f;   // 1/sqrt(512)
    stim[row] = fminf(fmaxf(n, 0.f), 1.f);
  }
}

// ---------------- cell recurrence (W == 0 => neighbors == 0) ----------------
__global__ void cell_kernel(const float* __restrict__ h_in,
                            const float* __restrict__ stim,
                            const float* aE, const float* aP, const float* aG,
                            const float* aL, const float* bP, const float* bG,
                            const float* bL,
                            unsigned short* __restrict__ inner) {
  int idx = blockIdx.x * blockDim.x + threadIdx.x;
  if (idx >= CD_ROWS * CD_NB) return;
  int row = idx >> 3, nb = idx & 7;
  float s = stim[row];
  const float* hp = h_in + (size_t)row * 32 + nb * 4;
  float E = hp[0], P = hp[1], G = hp[2], L = hp[3];
  float cAE = aE[0], cAP = aP[0], cAG = aG[0];
  float cBP = bP[0], cBG = bG[0], cBL = bL[0];
  (void)aL;  // alpha_L multiplies 'good' which is identically zero
#pragma unroll
  for (int st = 0; st < CD_STEPS; ++st) {
    float En = fminf(fmaxf(E + cAE * s - 0.4f * P - 0.2f * G, 0.f), 1.f);
    float Pn = fminf(fmaxf(P + cAP * s - cBP * P - 0.2f * E, 0.f), 1.f);
    float Gn = fminf(fmaxf(G + cAG * E * (1.f - P) - cBG * G - 0.3f * P, 0.f), 1.f);
    float Ln = fminf(fmaxf(L - cBL * L - 0.3f * P, 0.f), 1.f);
    E = En; P = Pn; G = Gn; L = Ln;
  }
  unsigned short* ip = inner + (size_t)row * CD_INNER;
  ip[nb * 4 + 0] = f2bf(E);
  ip[nb * 4 + 1] = f2bf(P);
  ip[nb * 4 + 2] = f2bf(G);
  ip[nb * 4 + 3] = f2bf(L);
#pragma unroll
  for (int j = 0; j < 8; ++j) ip[32 + nb * 8 + j] = 0;   // W block is zero
}

// ---------------- LayerNorm (one wave per row of 512) ----------------
__global__ void ln_kernel(const float* __restrict__ x,
                          const float* __restrict__ g,
                          const float* __restrict__ b,
                          float* __restrict__ outF,
                          unsigned short* __restrict__ outH) {
  int row  = blockIdx.x * 8 + (threadIdx.x >> 5);
  int lane = threadIdx.x & 31;
  const float* xr = x + (size_t)row * CD_D;
  float vals[16];
  float s = 0.f, s2 = 0.f;
#pragma unroll
  for (int i = 0; i < 16; ++i) {
    float v = xr[lane + i * 32];
    vals[i] = v; s += v; s2 += v * v;
  }
#pragma unroll
  for (int off = 16; off > 0; off >>= 1) {
    s  += __shfl_xor(s,  off, 32);
    s2 += __shfl_xor(s2, off, 32);
  }
  float mean = s * (1.f / CD_D);
  float var  = s2 * (1.f / CD_D) - mean * mean;
  float inv  = rsqrtf(var + 1e-5f);
#pragma unroll
  for (int i = 0; i < 16; ++i) {
    int d = lane + i * 32;
    float v = (vals[i] - mean) * inv * g[d] + b[d];
    if (outF) outF[(size_t)row * CD_D + d] = v;
    if (outH) outH[(size_t)row * CD_D + d] = f2bf(v);
  }
}

// ---------------- generic WMMA bf16 GEMM: C = A[MxK] * W[NxK]^T ----------------
// Each wave: 16x128 tile (8 accumulators). Column indices are clamped (not
// masked) for loads; out-of-range tiles accumulate garbage that the epilogue
// discards, so the K-loop is branch-free. Pointers are explicitly
// addrspace(1) so the pointer array still lowers to global_load (not FLAT),
// and all 18 fragment loads of a k-chunk are issued before the 8 WMMAs to
// maximize loads-in-flight.
// act: 0=none, 1=gelu(exact), 2=sigmoid.  res/outF/outH nullable.
__global__ void __launch_bounds__(256) gemm_bf16_kernel(
    const unsigned short* __restrict__ A, const unsigned short* __restrict__ W,
    const float* __restrict__ bias, const float* __restrict__ res,
    float* __restrict__ outF, unsigned short* __restrict__ outH,
    int M, int N, int K, int act) {
  int lane = threadIdx.x & 31;
  int wave = threadIdx.x >> 5;
  int l15 = lane & 15;
  int hi  = (lane & 16) ? 1 : 0;
  int rowBase = blockIdx.y * 128 + wave * 16;
  int colBase = blockIdx.x * 128;
  int arow = rowBase + l15;
  if (arow >= M) arow = M - 1;

  // A fragment: lane<16 -> K {kk..kk+7, kk+16..kk+23}; lane>=16 -> +8
  gusp ap = to_global(A) + (size_t)arow * K + hi * 8;
  // B fragment pointers (one per 16-wide tile), clamped columns
  gusp wp[8];
#pragma unroll
  for (int t = 0; t < 8; ++t) {
    int col = colBase + t * 16 + l15;
    if (col >= N) col = N - 1;
    wp[t] = to_global(W) + (size_t)col * K + hi * 8;
  }

  f32x8 acc[8];
#pragma unroll
  for (int t = 0; t < 8; ++t)
#pragma unroll
    for (int r = 0; r < 8; ++r) acc[t][r] = 0.f;

  for (int kk = 0; kk < K; kk += 32) {
    FragU UA, UB[8];
    load_frag(UA, ap);
    ap += 32;
#pragma unroll
    for (int t = 0; t < 8; ++t) {
      load_frag(UB[t], wp[t]);
      wp[t] += 32;
    }
#pragma unroll
    for (int t = 0; t < 8; ++t) {
      acc[t] = __builtin_amdgcn_wmma_f32_16x16x32_bf16(
          false, UA.v, false, UB[t].v, (short)0, acc[t], false, false);
    }
  }
#pragma unroll
  for (int t = 0; t < 8; ++t) {
    int col = colBase + t * 16 + l15;
    if (col >= N) continue;
    float bv = bias ? bias[col] : 0.f;
#pragma unroll
    for (int r = 0; r < 8; ++r) {
      int row = rowBase + r + hi * 8;
      if (row >= M) continue;
      float v = acc[t][r] + bv;
      if (act == 1)      v = 0.5f * v * (1.f + erff(v * 0.70710678118654752f));
      else if (act == 2) v = 1.f / (1.f + __expf(-v));
      size_t idx = (size_t)row * N + col;
      if (res)  v += res[idx];
      if (outF) outF[idx] = v;
      if (outH) outH[idx] = f2bf(v);
    }
  }
}

// ---------------- flash attention: one wave per 16-query tile ----------------
__global__ void __launch_bounds__(128) attn_kernel(
    const unsigned short* __restrict__ qkv, unsigned short* __restrict__ obuf) {
  const int ld = 3 * CD_D;   // 1536
  int lane = threadIdx.x & 31;
  int wave = threadIdx.x >> 5;
  int g  = blockIdx.x * 4 + wave;       // ((b*H + h) * S/16 + qt)
  int qt = g & (CD_S / 16 - 1);
  int bh = g >> 7;
  int hh = bh & (CD_H - 1);
  int bb = bh >> 3;
  int q0 = qt * 16;
  gusp base = to_global(qkv) + (size_t)bb * CD_S * ld;

  __shared__ float          sS[4][16 * 32];
  __shared__ unsigned short sP[4][16 * 32];
  __shared__ float          sCorr[4][16];
  __shared__ float          sStat[4][16];

  int l15 = lane & 15;
  int hi  = (lane & 16) ? 1 : 0;

  // Q fragments: 16x64, two 32-wide K chunks over DH
  bf16x16 aq[2];
  {
    gusp qp = base + (size_t)(q0 + l15) * ld + hh * CD_DH;
#pragma unroll
    for (int c = 0; c < 2; ++c) {
      FragU U;
      load_frag(U, qp + c * 32 + hi * 8);
      aq[c] = U.v;
    }
  }

  f32x8 o[4];
#pragma unroll
  for (int t = 0; t < 4; ++t)
#pragma unroll
    for (int r = 0; r < 8; ++r) o[t][r] = 0.f;

  float m = -1e30f, l = 0.f;
  int kend = q0 + 16;                      // causal: keys <= q0+15
  for (int kc = 0; kc < kend; kc += 32) {
    // ---- S = Q * K^T for 32 keys: batch all 4 K fragments, then 4 WMMAs ----
    FragU UK[2][2];
#pragma unroll
    for (int c = 0; c < 2; ++c)
#pragma unroll
      for (int t = 0; t < 2; ++t) {
        int key = kc + t * 16 + l15;
        load_frag(UK[c][t],
                  base + (size_t)key * ld + CD_D + hh * CD_DH + c * 32 + hi * 8);
      }
    f32x8 sc[2];
#pragma unroll
    for (int t = 0; t < 2; ++t)
#pragma unroll
      for (int r = 0; r < 8; ++r) sc[t][r] = 0.f;
#pragma unroll
    for (int c = 0; c < 2; ++c)
#pragma unroll
      for (int t = 0; t < 2; ++t)
        sc[t] = __builtin_amdgcn_wmma_f32_16x16x32_bf16(
            false, aq[c], false, UK[c][t].v, (short)0, sc[t], false, false);
    // scale + causal mask -> LDS scores
#pragma unroll
    for (int t = 0; t < 2; ++t)
#pragma unroll
      for (int r = 0; r < 8; ++r) {
        int row16 = r + hi * 8;
        int key = kc + t * 16 + l15;
        float v = sc[t][r] * 0.125f;       // 1/sqrt(DH)
        if (key > q0 + row16) v = -1e30f;
        sS[wave][row16 * 32 + t * 16 + l15] = v;
      }
    asm volatile("s_wait_dscnt 0x0" ::: "memory");
    // online softmax: lanes 0..15 each own one query row
    if (lane < 16) {
      float cmax = -1e30f;
#pragma unroll 4
      for (int j = 0; j < 32; ++j) cmax = fmaxf(cmax, sS[wave][lane * 32 + j]);
      float mnew = fmaxf(m, cmax);
      float corr = __expf(m - mnew);
      float ps = 0.f;
#pragma unroll 4
      for (int j = 0; j < 32; ++j) {
        float p = __expf(sS[wave][lane * 32 + j] - mnew);
        ps += p;
        sP[wave][lane * 32 + j] = f2bf(p);
      }
      l = l * corr + ps;
      m = mnew;
      sCorr[wave][lane] = corr;
    }
    asm volatile("s_wait_dscnt 0x0" ::: "memory");
    // ---- V fragments (strided): batch all 4 d-tiles, then rescale + WMMAs ----
    FragU UV[4];
#pragma unroll
    for (int t = 0; t < 4; ++t) {
      gusp vp = base + (size_t)(kc + hi * 16) * ld + 2 * CD_D +
                hh * CD_DH + t * 16 + l15;
#pragma unroll
      for (int j = 0; j < 16; ++j) UV[t].s[j] = vp[(size_t)j * ld];
    }
    // rescale O by per-row correction
    float cr[8];
#pragma unroll
    for (int r = 0; r < 8; ++r) cr[r] = sCorr[wave][r + hi * 8];
#pragma unroll
    for (int t = 0; t < 4; ++t)
#pragma unroll
      for (int r = 0; r < 8; ++r) o[t][r] *= cr[r];
    // P fragment from LDS (16x32 bf16, A layout)
    bf16x16 aP;
    {
      const unsigned short* pp = &sP[wave][l15 * 32 + hi * 16];
      FragU U;
      U.q[0] = *(const u32x4*)pp;
      U.q[1] = *(const u32x4*)(pp + 8);
      aP = U.v;
    }
#pragma unroll
    for (int t = 0; t < 4; ++t)
      o[t] = __builtin_amdgcn_wmma_f32_16x16x32_bf16(
          false, aP, false, UV[t].v, (short)0, o[t], false, false);
  }
  if (lane < 16) sStat[wave][lane] = 1.f / l;
  asm volatile("s_wait_dscnt 0x0" ::: "memory");
  float ir[8];
#pragma unroll
  for (int r = 0; r < 8; ++r) ir[r] = sStat[wave][r + hi * 8];
#pragma unroll
  for (int t = 0; t < 4; ++t)
#pragma unroll
    for (int r = 0; r < 8; ++r) {
      int row = q0 + r + hi * 8;
      int d = hh * CD_DH + t * 16 + l15;
      obuf[((size_t)bb * CD_S + row) * CD_D + d] = f2bf(o[t][r] * ir[r]);
    }
}

// ---------------- host driver ----------------
// Input order assumption: setup_inputs() insertion order, recursively flattened.
enum {
  IN_TOK = 0, IN_TOKEMB, IN_POSEMB, IN_HPW, IN_HPB,
  IN_AE, IN_AP, IN_AG, IN_AL, IN_BP, IN_BG, IN_BL, IN_WGROW, IN_WDECAY,
  IN_COLW, IN_COLB, IN_COLNG, IN_COLNB,
  IN_L0 = 18,                 // +12 per layer: in_w,in_b,out_w,out_b,w1,b1,w2,b2,n1g,n1b,n2g,n2b
  IN_FNG = 18 + 12 * CD_DEPTH,
  IN_FNB
};

extern "C" void kernel_launch(void* const* d_in, const int* in_sizes, int n_in,
                              void* d_out, int out_size, void* d_ws, size_t ws_size,
                              hipStream_t stream) {
  (void)in_sizes; (void)n_in; (void)out_size; (void)ws_size;
  char* ws = (char*)d_ws;
  size_t off = 0;
  auto take = [&](size_t bytes) -> char* {
    char* p = ws + off;
    off += (bytes + 255) & ~(size_t)255;
    return p;
  };
  const size_t R = CD_ROWS;
  float*          e_f32   = (float*)take(R * CD_D * 4);
  unsigned short* e_bf    = (unsigned short*)take(R * CD_D * 2);
  float*          stim    = (float*)take(R * 4);
  float*          h_f32   = (float*)take(R * 32 * 4);
  unsigned short* innerbf = (unsigned short*)take(R * CD_INNER * 2);
  float*          tmp_f32 = (float*)take(R * CD_D * 4);
  float*          x_f32   = (float*)take(R * CD_D * 4);
  unsigned short* xn_bf   = (unsigned short*)take(R * CD_D * 2);
  unsigned short* qkv_bf  = (unsigned short*)take(R * 3 * CD_D * 2);
  unsigned short* o_bf    = (unsigned short*)take(R * CD_D * 2);
  unsigned short* ff_bf   = (unsigned short*)take(R * CD_FF * 2);
  unsigned short* xf_bf   = (unsigned short*)take(R * CD_D * 2);
  unsigned short* wtok_bf = (unsigned short*)take((size_t)CD_V * CD_D * 2);
  unsigned short* whp_bf  = (unsigned short*)take(32 * CD_D * 2);
  unsigned short* wcol_bf = (unsigned short*)take(CD_D * CD_INNER * 2);
  unsigned short* win_bf[CD_DEPTH];
  unsigned short* wout_bf[CD_DEPTH];
  unsigned short* w1_bf[CD_DEPTH];
  unsigned short* w2_bf[CD_DEPTH];
  for (int L = 0; L < CD_DEPTH; ++L) {
    win_bf[L]  = (unsigned short*)take((size_t)3 * CD_D * CD_D * 2);
    wout_bf[L] = (unsigned short*)take((size_t)CD_D * CD_D * 2);
    w1_bf[L]   = (unsigned short*)take((size_t)CD_FF * CD_D * 2);
    w2_bf[L]   = (unsigned short*)take((size_t)CD_D * CD_FF * 2);
  }

  auto conv = [&](int idx, unsigned short* dst, long n) {
    int blocks = (int)((n + 1023) / 1024);
    if (blocks > 4096) blocks = 4096;
    f2bf_kernel<<<blocks, 256, 0, stream>>>((const float*)d_in[idx], dst, n);
  };
  auto gemm = [&](const unsigned short* A, const unsigned short* W,
                  const float* bias, const float* res, float* outF,
                  unsigned short* outH, int N, int K, int act) {
    dim3 grid((N + 127) / 128, (CD_ROWS + 127) / 128);
    gemm_bf16_kernel<<<grid, 256, 0, stream>>>(A, W, bias, res, outF, outH,
                                               CD_ROWS, N, K, act);
  };

  // weight conversions
  conv(IN_TOKEMB, wtok_bf, (long)CD_V * CD_D);
  conv(IN_HPW,    whp_bf,  32L * CD_D);
  conv(IN_COLW,   wcol_bf, (long)CD_D * CD_INNER);
  for (int L = 0; L < CD_DEPTH; ++L) {
    conv(IN_L0 + 12 * L + 0, win_bf[L],  3L * CD_D * CD_D);
    conv(IN_L0 + 12 * L + 2, wout_bf[L], (long)CD_D * CD_D);
    conv(IN_L0 + 12 * L + 4, w1_bf[L],   (long)CD_FF * CD_D);
    conv(IN_L0 + 12 * L + 6, w2_bf[L],   (long)CD_D * CD_FF);
  }

  // embedding + stim
  embed_kernel<<<CD_ROWS, 256, 0, stream>>>(
      (const int*)d_in[IN_TOK], (const float*)d_in[IN_TOKEMB],
      (const float*)d_in[IN_POSEMB], e_f32, e_bf, stim);

  // h = sigmoid(e @ h_proj_w.T + h_proj_b)
  gemm(e_bf, whp_bf, (const float*)d_in[IN_HPB], nullptr, h_f32, nullptr,
       32, CD_D, 2);

  // cell recurrence -> inner (bf16), W-part zero
  cell_kernel<<<(CD_ROWS * CD_NB + 255) / 256, 256, 0, stream>>>(
      h_f32, stim,
      (const float*)d_in[IN_AE], (const float*)d_in[IN_AP],
      (const float*)d_in[IN_AG], (const float*)d_in[IN_AL],
      (const float*)d_in[IN_BP], (const float*)d_in[IN_BG],
      (const float*)d_in[IN_BL], innerbf);

  // tmp = e + inner @ col_w.T + col_b ; x = LN(tmp)
  gemm(innerbf, wcol_bf, (const float*)d_in[IN_COLB], e_f32, tmp_f32, nullptr,
       CD_D, CD_INNER, 0);
  ln_kernel<<<CD_ROWS / 8, 256, 0, stream>>>(
      tmp_f32, (const float*)d_in[IN_COLNG], (const float*)d_in[IN_COLNB],
      x_f32, nullptr);

  // transformer blocks
  for (int L = 0; L < CD_DEPTH; ++L) {
    const float* in_b  = (const float*)d_in[IN_L0 + 12 * L + 1];
    const float* out_b = (const float*)d_in[IN_L0 + 12 * L + 3];
    const float* b1    = (const float*)d_in[IN_L0 + 12 * L + 5];
    const float* b2    = (const float*)d_in[IN_L0 + 12 * L + 7];
    const float* n1g   = (const float*)d_in[IN_L0 + 12 * L + 8];
    const float* n1b   = (const float*)d_in[IN_L0 + 12 * L + 9];
    const float* n2g   = (const float*)d_in[IN_L0 + 12 * L + 10];
    const float* n2b   = (const float*)d_in[IN_L0 + 12 * L + 11];

    ln_kernel<<<CD_ROWS / 8, 256, 0, stream>>>(x_f32, n1g, n1b, nullptr, xn_bf);
    gemm(xn_bf, win_bf[L], in_b, nullptr, nullptr, qkv_bf, 3 * CD_D, CD_D, 0);
    attn_kernel<<<(CD_B * CD_H * (CD_S / 16)) / 4, 128, 0, stream>>>(qkv_bf, o_bf);
    gemm(o_bf, wout_bf[L], out_b, x_f32, x_f32, nullptr, CD_D, CD_D, 0);
    ln_kernel<<<CD_ROWS / 8, 256, 0, stream>>>(x_f32, n2g, n2b, nullptr, xn_bf);
    gemm(xn_bf, w1_bf[L], b1, nullptr, nullptr, ff_bf, CD_FF, CD_D, 1);
    gemm(ff_bf, w2_bf[L], b2, x_f32, x_f32, nullptr, CD_D, CD_FF, 0);
  }

  // final LN + tied LM head
  ln_kernel<<<CD_ROWS / 8, 256, 0, stream>>>(
      x_f32, (const float*)d_in[IN_FNG], (const float*)d_in[IN_FNB],
      nullptr, xf_bf);
  gemm(xf_bf, wtok_bf, nullptr, nullptr, (float*)d_out, nullptr, CD_V, CD_D, 0);
}
